// MultiHeadAttention_78245714198593
// MI455X (gfx1250) — matrix operand order repaired and checked by reading, not compiled, not run on hardware
//
#include <hip/hip_runtime.h>
#include <hip/hip_bf16.h>
#include <stdint.h>

// ---------------------------------------------------------------------------
// CDNA5 (gfx1250) types
// ---------------------------------------------------------------------------
typedef __attribute__((ext_vector_type(16))) __bf16 v16bf;
typedef __attribute__((ext_vector_type(8)))  __bf16 v8bf;
typedef __attribute__((ext_vector_type(8)))  float  v8f;
typedef __attribute__((ext_vector_type(8)))  float  f32x8;
typedef __attribute__((ext_vector_type(4)))  unsigned int u32x4;
typedef __attribute__((ext_vector_type(8)))  int    i32x8;
typedef __attribute__((ext_vector_type(4)))  int    i32x4;

#define DEVINL __device__ __forceinline__

constexpr int NB = 16;       // batch
constexpr int NT = 512;      // query tokens
constexpr int NC = 512;      // key/value tokens
constexpr int E  = 512;      // embed
constexpr int H  = 8;        // heads
constexpr int DK = 64;
constexpr int DV = 64;
constexpr int HD = H * DK;   // 512
constexpr int LSTR = 72;     // LDS tile row stride in bf16 (64 data + 4-DWORD TDM pad)

// ---------------------------------------------------------------------------
// WMMA fragment helpers (layouts per CDNA5 ISA §7.12.2, wave32)
//  A (16x32 bf16): lane L, row M=L%16; halves 0-7 = K k0+{0..7}+8*(L>=16),
//                  halves 8-15 = K k0+16+{0..7}+8*(L>=16)
//  B (32x16 bf16): lane L, col N=L%16; same K order as A
//  C/D (16x16 f32): vgpr r -> row r+8*(L>=16), col L%16
// ---------------------------------------------------------------------------
DEVINL v16bf load_a_bf(const __bf16* A, int lda, int m, int k0, int l15, int lhi) {
    const __bf16* p = A + (size_t)(m + l15) * lda + (k0 + lhi * 8);
    union { v16bf v; v8bf h[2]; } u;
    u.h[0] = *(const v8bf*)(p);
    u.h[1] = *(const v8bf*)(p + 16);
    return u.v;
}

DEVINL v16bf load_a_f32(const float* A, int lda, int m, int k0, int l15, int lhi) {
    const float* p = A + (size_t)(m + l15) * lda + (k0 + lhi * 8);
    f32x8 lo = *(const f32x8*)(p);
    f32x8 hi = *(const f32x8*)(p + 16);
    v16bf r;
#pragma unroll
    for (int i = 0; i < 8; ++i) { r[i] = (__bf16)lo[i]; r[8 + i] = (__bf16)hi[i]; }
    return r;
}

DEVINL v16bf load_b_bf(const __bf16* Bm, int ldb, int n, int k0, int l15, int lhi) {
    const __bf16* p = Bm + (size_t)(n + l15) * ldb + (k0 + lhi * 8);
    union { v16bf v; v8bf h[2]; } u;
    u.h[0] = *(const v8bf*)(p);
    u.h[1] = *(const v8bf*)(p + 16);
    return u.v;
}

// LDS-tile fragment loads (row stride 72 bf16 = 144B -> conflict-free ds_load)
DEVINL v16bf lds_frag_a(const __bf16* T, int r0, int kk, int l15, int lhi) {
    const __bf16* p = T + (r0 + l15) * LSTR + kk + lhi * 8;
    union { v16bf v; v8bf h[2]; } u;
    u.h[0] = *(const v8bf*)(p);
    u.h[1] = *(const v8bf*)(p + 16);
    return u.v;
}
DEVINL v16bf lds_frag_b(const __bf16* T, int r0, int kk, int l15, int lhi) {
    const __bf16* p = T + (r0 + l15) * LSTR + kk + lhi * 8;
    union { v16bf v; v8bf h[2]; } u;
    u.h[0] = *(const v8bf*)(p);
    u.h[1] = *(const v8bf*)(p + 16);
    return u.v;
}

DEVINL v8f wmma_bf16(v16bf a, v16bf b, v8f c) {
    return __builtin_amdgcn_wmma_f32_16x16x32_bf16(false, a, false, b, (short)0, c, false, false);
}

// ---------------------------------------------------------------------------
// Tensor Data Mover: DMA one 64-row x 64-col bf16 tile (row stride `ld` elems)
// into LDS, padding each 128B row by 16B so LDS rows land on a 144B stride.
// D# packing per CDNA5 ISA (groups 0/1; 2D tile).
// ---------------------------------------------------------------------------
DEVINL void tdm_load_tile(const __bf16* src, int ld, int row0, int k0, int rowsTotal,
                          __bf16* lds_dst) {
    const uint64_t gaddr = (uint64_t)(uintptr_t)(src + (size_t)row0 * ld + k0);
    const uint32_t laddr = (uint32_t)(uintptr_t)lds_dst;   // LDS byte address
    u32x4 g0;
    g0.x = 1u;                                             // count=1 valid descriptor
    g0.y = laddr;                                          // lds_addr [63:32]
    g0.z = (uint32_t)gaddr;                                // global_addr lo
    g0.w = (uint32_t)((gaddr >> 32) & 0x1FFFFFFu) | (2u << 30);  // addr hi | type=2
    i32x8 g1;
    // data_size=2B(1), pad_enable=1, pad_interval=32 DWORDs(code 4), pad_amount=4 DWORDs(code 3)
    g1[0] = (1 << 16) | (1 << 20) | (4 << 22) | (3 << 25);
    g1[1] = (int)(((uint32_t)ld & 0xFFFFu) << 16);                       // tensor_dim0 lo16
    g1[2] = (int)(((uint32_t)ld >> 16) | (((uint32_t)rowsTotal & 0xFFFFu) << 16)); // dim0 hi / dim1 lo
    g1[3] = (int)(((uint32_t)rowsTotal >> 16) | (64u << 16));            // dim1 hi | tile_dim0=64
    g1[4] = 64;                                                          // tile_dim1=64, tile_dim2=0
    g1[5] = ld;                                                          // tensor_dim0_stride lo32
    g1[6] = 0;                                                           // stride0 hi / stride1 lo
    g1[7] = 0;
    i32x4 z4 = {};
#if defined(__clang_major__) && (__clang_major__ >= 23)
    i32x8 z8 = {};
    __builtin_amdgcn_tensor_load_to_lds(g0, g1, z4, z4, z8, 0);
#else
    __builtin_amdgcn_tensor_load_to_lds(g0, g1, z4, z4, 0);
#endif
}

// ---------------------------------------------------------------------------
// f32 -> bf16 conversion
// ---------------------------------------------------------------------------
__global__ __launch_bounds__(256) void cvt_kernel(const float* __restrict__ src,
                                                  __bf16* __restrict__ dst, int n) {
    int i = blockIdx.x * 256 + threadIdx.x;
    if (i < n) dst[i] = (__bf16)src[i];
}

// ---------------------------------------------------------------------------
// TDM-staged GEMM: C = A * Bsrc^T + bias.  A:[M x Kd], Bsrc:[N x Kd] bf16
// row-major.  Block = 128 thr (4 waves 2x2), block tile 64x64, K chunk 64,
// double-buffered LDS tiles filled by the Tensor Data Mover (wave 0 issues,
// TENSORcnt-gated, workgroup barrier hands tiles to all 4 waves).
// OUT: 0 = f32 row-major, 1 = bf16 row-major, 2 = bf16 transposed Vt layout.
// ---------------------------------------------------------------------------
template <int OUT>
__global__ __launch_bounds__(128) void gemm_tdm_kernel(const __bf16* __restrict__ A,
                                                       const __bf16* __restrict__ Bm,
                                                       const float* __restrict__ bias,
                                                       void* __restrict__ Cout,
                                                       int M, int N, int Kd) {
    __shared__ __bf16 As[2][64 * LSTR];
    __shared__ __bf16 Bs[2][64 * LSTR];
    const int lane = threadIdx.x & 31, l15 = lane & 15, lhi = lane >> 4;
    const int wave = threadIdx.x >> 5;
    const int mblk = blockIdx.x * 64, nblk = blockIdx.y * 64;
    const int mw = (wave & 1) * 32, nw = (wave >> 1) * 32;
    const int nchunks = Kd >> 6;

    if (wave == 0) {                      // wave-uniform guard (TDM ignores EXEC)
        tdm_load_tile(A,  Kd, mblk, 0, M, As[0]);
        tdm_load_tile(Bm, Kd, nblk, 0, N, Bs[0]);
    }
    v8f acc[2][2] = {};
    for (int ch = 0; ch < nchunks; ++ch) {
        const int p = ch & 1;
        if (wave == 0) {
            if (ch + 1 < nchunks) {
                tdm_load_tile(A,  Kd, mblk, (ch + 1) << 6, M, As[p ^ 1]);
                tdm_load_tile(Bm, Kd, nblk, (ch + 1) << 6, N, Bs[p ^ 1]);
                __builtin_amdgcn_s_wait_tensorcnt(2);   // chunk ch landed
            } else {
                __builtin_amdgcn_s_wait_tensorcnt(0);
            }
        }
        __syncthreads();                  // tiles visible to all waves
        const __bf16* At = As[p];
        const __bf16* Bt = Bs[p];
#pragma unroll
        for (int kk = 0; kk < 64; kk += 32) {
            v16bf a0 = lds_frag_a(At, mw,      kk, l15, lhi);
            v16bf a1 = lds_frag_a(At, mw + 16, kk, l15, lhi);
            v16bf b0 = lds_frag_b(Bt, nw,      kk, l15, lhi);
            v16bf b1 = lds_frag_b(Bt, nw + 16, kk, l15, lhi);
            acc[0][0] = wmma_bf16(a0, b0, acc[0][0]);
            acc[0][1] = wmma_bf16(a0, b1, acc[0][1]);
            acc[1][0] = wmma_bf16(a1, b0, acc[1][0]);
            acc[1][1] = wmma_bf16(a1, b1, acc[1][1]);
        }
        __syncthreads();                  // all reads done before buffer reuse
    }
#pragma unroll
    for (int i = 0; i < 2; ++i)
#pragma unroll
        for (int j = 0; j < 2; ++j) {
            const int nn = nblk + nw + j * 16 + l15;
            const float bv = bias[nn];
#pragma unroll
            for (int r = 0; r < 8; ++r) {
                const int mm = mblk + mw + i * 16 + lhi * 8 + r;
                const float v = acc[i][j][r] + bv;
                if (OUT == 0) {
                    ((float*)Cout)[(size_t)mm * N + nn] = v;
                } else if (OUT == 1) {
                    ((__bf16*)Cout)[(size_t)mm * N + nn] = (__bf16)v;
                } else {                   // Vt[b][h][d][c], mm = b*NC+c, nn = h*64+d
                    const int b = mm >> 9, c = mm & 511;
                    const int h = nn >> 6, dd = nn & 63;
                    ((__bf16*)Cout)[(((size_t)(b * H + h)) * DV + dd) * NC + c] = (__bf16)v;
                }
            }
        }
}

// ---------------------------------------------------------------------------
// Per-(b,h,row) L2 norm over 64 bf16 channels. One wave32 per norm.
// ---------------------------------------------------------------------------
__global__ __launch_bounds__(256) void rownorm_kernel(const __bf16* __restrict__ X,
                                                      float* __restrict__ out, int Nr) {
    const int w = blockIdx.x * 8 + (threadIdx.x >> 5);
    const int lane = threadIdx.x & 31;
    const int h = w & (H - 1);
    const int row = w >> 3;                    // b*Nr + t
    const __bf16* p = X + (size_t)row * HD + h * DK + lane * 2;
    const float a = (float)p[0], b2 = (float)p[1];
    float s = a * a + b2 * b2;
#pragma unroll
    for (int d = 16; d; d >>= 1) s += __shfl_xor(s, d, 32);
    if (lane == 0) {
        const int b = row / Nr, t = row % Nr;
        out[((size_t)(b * H + h)) * Nr + t] = sqrtf(s);
    }
}

// ---------------------------------------------------------------------------
// Scores: per (b,h), S = Qh * Kh^T (K=64, direct-global WMMA), fused cosine
// normalization (v_rcp_f32 instead of IEEE div chain) + pos_bias + mask.
// Raw scores land in d_out's att region.
// ---------------------------------------------------------------------------
__global__ __launch_bounds__(128) void scores_kernel(const __bf16* __restrict__ Qp,
                                                     const __bf16* __restrict__ Kp,
                                                     const float* __restrict__ qn,
                                                     const float* __restrict__ kn,
                                                     const float* __restrict__ pos_bias,
                                                     const int* __restrict__ mask,
                                                     float* __restrict__ attOut) {
    const int bh = blockIdx.z, b = bh >> 3, h = bh & 7;
    const __bf16* Ab = Qp + (size_t)b * NT * HD + h * DK;   // lda = HD
    const __bf16* Bb = Kp + (size_t)b * NC * HD + h * DK;   // ldb = HD
    const int lane = threadIdx.x & 31, l15 = lane & 15, lhi = lane >> 4;
    const int wave = threadIdx.x >> 5;
    const int m0 = blockIdx.x * 64 + (wave & 1) * 32;
    const int n0 = blockIdx.y * 64 + (wave >> 1) * 32;
    v8f acc[2][2] = {};
#pragma unroll
    for (int k0 = 0; k0 < DK; k0 += 32) {
        v16bf a0 = load_a_bf(Ab, HD, m0,      k0, l15, lhi);
        v16bf a1 = load_a_bf(Ab, HD, m0 + 16, k0, l15, lhi);
        v16bf b0 = load_b_bf(Bb, HD, n0,      k0, l15, lhi);
        v16bf b1 = load_b_bf(Bb, HD, n0 + 16, k0, l15, lhi);
        acc[0][0] = wmma_bf16(a0, b0, acc[0][0]);
        acc[0][1] = wmma_bf16(a0, b1, acc[0][1]);
        acc[1][0] = wmma_bf16(a1, b0, acc[1][0]);
        acc[1][1] = wmma_bf16(a1, b1, acc[1][1]);
    }
    float* out = attOut + (size_t)bh * NT * NC;
    const float* qnr = qn + (size_t)bh * NT;
    const float* knr = kn + (size_t)bh * NC;
#pragma unroll
    for (int i = 0; i < 2; ++i)
#pragma unroll
        for (int j = 0; j < 2; ++j) {
            const int c = n0 + j * 16 + l15;
            const float kv = knr[c];
            const float pb = pos_bias[c];
            const bool mk = mask[b * NC + c] != 0;
#pragma unroll
            for (int r = 0; r < 8; ++r) {
                const int t = m0 + i * 16 + lhi * 8 + r;
                const float inv = __builtin_amdgcn_rcpf(fmaxf(qnr[t] * kv, 1e-6f));
                const float v = acc[i][j][r] * inv + pb;
                out[(size_t)t * NC + c] = mk ? -__builtin_inff() : v;
            }
        }
}

// ---------------------------------------------------------------------------
// In-place row softmax over 512 scores; one wave32 per row (16 vals/lane).
// ---------------------------------------------------------------------------
__global__ __launch_bounds__(256) void softmax_kernel(float* __restrict__ att) {
    const int w = blockIdx.x * 8 + (threadIdx.x >> 5);
    const int lane = threadIdx.x & 31;
    float* row = att + (size_t)w * NC;
    float x[16];
    float mx = -__builtin_inff();
#pragma unroll
    for (int j = 0; j < 16; ++j) { x[j] = row[j * 32 + lane]; mx = fmaxf(mx, x[j]); }
#pragma unroll
    for (int d = 16; d; d >>= 1) mx = fmaxf(mx, __shfl_xor(mx, d, 32));
    float sum = 0.f;
#pragma unroll
    for (int j = 0; j < 16; ++j) { x[j] = __expf(x[j] - mx); sum += x[j]; }
#pragma unroll
    for (int d = 16; d; d >>= 1) sum += __shfl_xor(sum, d, 32);
    const float inv = __builtin_amdgcn_rcpf(sum);
#pragma unroll
    for (int j = 0; j < 16; ++j) row[j * 32 + lane] = x[j] * inv;
}

// ---------------------------------------------------------------------------
// P*V: per (b,h), Oh = att(f32, converted on the fly) * Vt^T  -> bf16 heads.
// ---------------------------------------------------------------------------
__global__ __launch_bounds__(128) void pv_kernel(const float* __restrict__ att,
                                                 const __bf16* __restrict__ Vt,
                                                 __bf16* __restrict__ Oh) {
    const int bh = blockIdx.z, b = bh >> 3, h = bh & 7;
    const float*  Ab = att + (size_t)bh * NT * NC;
    const __bf16* Bb = Vt  + (size_t)bh * DV * NC;
    const int lane = threadIdx.x & 31, l15 = lane & 15, lhi = lane >> 4;
    const int wave = threadIdx.x >> 5;
    const int m0 = blockIdx.x * 64 + (wave & 1) * 32;
    const int n0 = (wave >> 1) * 32;           // N = DV = 64
    v8f acc[2][2] = {};
    for (int k0 = 0; k0 < NC; k0 += 32) {
        v16bf a0 = load_a_f32(Ab, NC, m0,      k0, l15, lhi);
        v16bf a1 = load_a_f32(Ab, NC, m0 + 16, k0, l15, lhi);
        v16bf b0 = load_b_bf(Bb, NC, n0,      k0, l15, lhi);
        v16bf b1 = load_b_bf(Bb, NC, n0 + 16, k0, l15, lhi);
        acc[0][0] = wmma_bf16(a0, b0, acc[0][0]);
        acc[0][1] = wmma_bf16(a0, b1, acc[0][1]);
        acc[1][0] = wmma_bf16(a1, b0, acc[1][0]);
        acc[1][1] = wmma_bf16(a1, b1, acc[1][1]);
    }
#pragma unroll
    for (int i = 0; i < 2; ++i)
#pragma unroll
        for (int j = 0; j < 2; ++j) {
            const int nn = n0 + j * 16 + l15;
#pragma unroll
            for (int r = 0; r < 8; ++r) {
                const int mm = m0 + i * 16 + lhi * 8 + r;
                Oh[(size_t)(b * NT + mm) * HD + h * DV + nn] = (__bf16)acc[i][j][r];
            }
        }
}

// ---------------------------------------------------------------------------
// Host-side orchestration
// ---------------------------------------------------------------------------
extern "C" void kernel_launch(void* const* d_in, const int* in_sizes, int n_in,
                              void* d_out, int out_size, void* d_ws, size_t ws_size,
                              hipStream_t stream) {
    const float* queries = (const float*)d_in[0];
    const float* keys    = (const float*)d_in[1];
    const float* values  = (const float*)d_in[2];
    const float* At_w    = (const float*)d_in[3];
    const float* At_b    = (const float*)d_in[4];
    const float* Ac_w    = (const float*)d_in[5];
    const float* Ac_b    = (const float*)d_in[6];
    const float* Bc_w    = (const float*)d_in[7];
    const float* Bc_b    = (const float*)d_in[8];
    const float* posb    = (const float*)d_in[9];
    const float* R_w     = (const float*)d_in[10];
    const float* R_b     = (const float*)d_in[11];
    const int*   mask    = (const int*)d_in[12];

    float* outMat = (float*)d_out;                              // [NB*NT, E]
    float* attOut = (float*)d_out + (size_t)NB * NT * E;        // [NB*H, NT, NC]

    // workspace carve (256B aligned slices)
    char* w = (char*)d_ws;
    auto carve = [&](size_t bytes) { void* p = (void*)w; w += (bytes + 255) & ~(size_t)255; return p; };
    const size_t ACT = (size_t)NB * 512 * 512;
    const size_t WEL = (size_t)512 * 512;
    __bf16* qbf = (__bf16*)carve(ACT * 2);
    __bf16* kbf = (__bf16*)carve(ACT * 2);
    __bf16* vbf = (__bf16*)carve(ACT * 2);
    __bf16* wq  = (__bf16*)carve(WEL * 2);
    __bf16* wk  = (__bf16*)carve(WEL * 2);
    __bf16* wv  = (__bf16*)carve(WEL * 2);
    __bf16* wr  = (__bf16*)carve(WEL * 2);
    __bf16* Qp  = (__bf16*)carve(ACT * 2);
    __bf16* Kp  = (__bf16*)carve(ACT * 2);
    __bf16* Vt  = (__bf16*)carve(ACT * 2);       // [NB,H,DV,NC]
    __bf16* Oh  = (__bf16*)carve(ACT * 2);       // [NB*NT, H*DV]
    float*  qn  = (float*)carve((size_t)NB * H * NT * 4);
    float*  kn  = (float*)carve((size_t)NB * H * NC * 4);

    auto cv = [&](const float* s, __bf16* d, int n) {
        cvt_kernel<<<dim3((n + 255) / 256), dim3(256), 0, stream>>>(s, d, n);
    };
    cv(queries, qbf, (int)ACT);
    cv(keys,    kbf, (int)ACT);
    cv(values,  vbf, (int)ACT);
    cv(At_w, wq, (int)WEL);
    cv(Ac_w, wk, (int)WEL);
    cv(Bc_w, wv, (int)WEL);
    cv(R_w,  wr, (int)WEL);

    const dim3 blk128(128), blk256(256);
    const dim3 gProj(NB * NT / 64, HD / 64);     // (128, 8)

    // Q / K projections (bf16 out), V projection (transposed bf16 out) — TDM-staged
    gemm_tdm_kernel<1><<<gProj, blk128, 0, stream>>>(qbf, wq, At_b, (void*)Qp, NB * NT, HD, E);
    gemm_tdm_kernel<1><<<gProj, blk128, 0, stream>>>(kbf, wk, Ac_b, (void*)Kp, NB * NC, HD, E);
    gemm_tdm_kernel<2><<<gProj, blk128, 0, stream>>>(vbf, wv, Bc_b, (void*)Vt, NB * NC, HD, E);

    // per-(b,h,row) norms
    rownorm_kernel<<<dim3(NB * NT * H / 8), blk256, 0, stream>>>(Qp, qn, NT);
    rownorm_kernel<<<dim3(NB * NC * H / 8), blk256, 0, stream>>>(Kp, kn, NC);

    // raw scores -> d_out att region, then softmax in place
    scores_kernel<<<dim3(NT / 64, NC / 64, NB * H), blk128, 0, stream>>>(
        Qp, Kp, qn, kn, posb, mask, attOut);
    softmax_kernel<<<dim3(NB * H * NT / 8), blk256, 0, stream>>>(attOut);

    // P*V (converts f32 att to bf16 fragments on the fly)
    pv_kernel<<<dim3(NT / 64, 1, NB * H), blk128, 0, stream>>>(attOut, Vt, Oh);

    // final projection: out = Oh @ R_w^T + R_b  (f32 out) — TDM-staged
    gemm_tdm_kernel<0><<<gProj, blk128, 0, stream>>>(Oh, wr, R_b, (void*)outMat, NB * NT, E, HD);
}
